// ThreeBodyInteraction_20770461843457
// MI455X (gfx1250) — matrix operand order repaired and checked by reading, not compiled, not run on hardware
//
#include <hip/hip_runtime.h>

#define E_EDGES 200000
#define T_TRIP  1000000
#define DF      128
#define NBF     20

typedef __attribute__((ext_vector_type(16))) __bf16 v16bf;
typedef __attribute__((ext_vector_type(8)))  __bf16 v8bf;
typedef __attribute__((ext_vector_type(4)))  __bf16 v4bf;
typedef __attribute__((ext_vector_type(8)))  float  v8f;

#define DELTA_BYTES ((size_t)E_EDGES * DF * 4)   // f32 accumulator
#define ATTR_BYTES  ((size_t)E_EDGES * DF * 2)   // bf16 edge_attr
#define ANG_BYTES   ((size_t)T_TRIP * 32 * 2)    // bf16 angle feats, padded 20->32
#define W1F_ELEMS   (9 * 8 * 32 * 16)            // 288x128 in B-frag layout
#define W2F_ELEMS   (4 * 8 * 32 * 16)            // 128x128 in B-frag layout

#if defined(__HIP_DEVICE_COMPILE__) && \
    __has_builtin(__builtin_amdgcn_global_load_async_to_lds_b128)
#define ASYNC_LDS_COPY 1
typedef int async_v4i __attribute__((vector_size(16)));
#endif

// Cooperative global -> LDS copy of `bytes` (multiple of 16) using the CDNA5
// async-LDS path when available; falls back to VGPR round-trip copies.
__device__ __forceinline__ void stage_to_lds(const void* src, void* dst,
                                             int bytes, int tid, int nthreads) {
#ifdef ASYNC_LDS_COPY
  char* s = (char*)src;
  char* d = (char*)dst;
  for (int i = tid; i < bytes / 16; i += nthreads) {
    __builtin_amdgcn_global_load_async_to_lds_b128(
        (async_v4i*)(s + (size_t)i * 16), (async_v4i*)(d + (size_t)i * 16), 0, 0);
  }
  asm volatile("s_wait_asynccnt 0" ::: "memory");
#else
  const int4* s = (const int4*)src;
  int4* d = (int4*)dst;
  for (int i = tid; i < bytes / 16; i += nthreads) d[i] = s[i];
#endif
}

// ---------------------------------------------------------------------------
// f32 -> bf16 convert of edge_attr (vectorized x4)
__global__ void cvt_attr_kernel(const float* __restrict__ a,
                                __bf16* __restrict__ o, int n4) {
  int i = blockIdx.x * blockDim.x + threadIdx.x;
  if (i < n4) {
    const float4 v = ((const float4*)a)[i];
    v4bf t = {(__bf16)v.x, (__bf16)v.y, (__bf16)v.z, (__bf16)v.w};
    ((v4bf*)o)[i] = t;
  }
}

// ---------------------------------------------------------------------------
// Pre-swizzle a (krows x 128) f32 weight into CDNA5 bf16 B-fragment order:
// frag[kt][nt][lane][j]  with  k = kt*32 + (lane>>4)*8 + (j&7) + (j>>3)*16,  n = nt*16 + (lane&15)
// Rows k >= krows are zero padding.
__global__ void build_wf_kernel(const float* __restrict__ W,
                                __bf16* __restrict__ wf,
                                int ktiles, int krows) {
  int idx = blockIdx.x * blockDim.x + threadIdx.x;
  int total = ktiles * 8 * 32 * 16;
  if (idx >= total) return;
  int j    = idx & 15;
  int lane = (idx >> 4) & 31;
  int nt   = (idx >> 9) & 7;
  int kt   = idx >> 12;
  int k = kt * 32 + ((lane >> 4) << 3) + (j & 7) + ((j >> 3) << 4);
  int n = (nt << 4) + (lane & 15);
  wf[idx] = (__bf16)((k < krows) ? W[k * DF + n] : 0.0f);
}

// ---------------------------------------------------------------------------
// Angle-feature MLP: one thread per triplet, output bf16[32] (20 feats + pad)
__global__ void angle_kernel(const int* __restrict__ tbe,
                             const float* __restrict__ ev,
                             const float* __restrict__ A1,
                             const float* __restrict__ ab1,
                             const float* __restrict__ A2,
                             const float* __restrict__ ab2,
                             __bf16* __restrict__ ang) {
  __shared__ float sA1[3 * NBF], sab1[NBF], sA2[NBF * NBF], sab2[NBF];
  for (int i = threadIdx.x; i < 3 * NBF; i += blockDim.x) sA1[i] = A1[i];
  for (int i = threadIdx.x; i < NBF; i += blockDim.x) { sab1[i] = ab1[i]; sab2[i] = ab2[i]; }
  for (int i = threadIdx.x; i < NBF * NBF; i += blockDim.x) sA2[i] = A2[i];
  __syncthreads();

  int t = blockIdx.x * blockDim.x + threadIdx.x;
  if (t >= T_TRIP) return;
  int eij = tbe[2 * t];
  int eik = tbe[2 * t + 1];
  float ax = ev[eij * 3 + 0], ay = ev[eij * 3 + 1], az = ev[eij * 3 + 2];
  float bx = ev[eik * 3 + 0], by = ev[eik * 3 + 1], bz = ev[eik * 3 + 2];
  // negation of both vectors cancels in norms and dot product
  float dij = fmaxf(sqrtf(ax * ax + ay * ay + az * az), 1e-6f);
  float dik = fmaxf(sqrtf(bx * bx + by * by + bz * bz), 1e-6f);
  float cs = (ax * bx + ay * by + az * bz) / (dij * dik);
  cs = fminf(fmaxf(cs, -1.0f), 1.0f);

  float hid[NBF];
#pragma unroll
  for (int j = 0; j < NBF; ++j) {
    float x = dij * sA1[j] + dik * sA1[NBF + j] + cs * sA1[2 * NBF + j] + sab1[j];
    hid[j] = x / (1.0f + __expf(-x));   // silu
  }
  __bf16 outv[32];
#pragma unroll
  for (int j = 0; j < NBF; ++j) {
    float acc = sab2[j];
#pragma unroll
    for (int h = 0; h < NBF; ++h) acc += hid[h] * sA2[h * NBF + j];
    outv[j] = (__bf16)acc;
  }
#pragma unroll
  for (int j = NBF; j < 32; ++j) outv[j] = (__bf16)0.0f;

  v4bf* dst = (v4bf*)(ang + (size_t)t * 32);
#pragma unroll
  for (int q = 0; q < 8; ++q) {
    v4bf p = {outv[4 * q], outv[4 * q + 1], outv[4 * q + 2], outv[4 * q + 3]};
    dst[q] = p;
  }
}

// ---------------------------------------------------------------------------
// Main fused kernel: per wave, a 16-triplet tile.
//   H = silu([attr[eij], attr[eik], ang] @ W1 + b1)   (16x128, WMMA over K=288)
//   M = H @ W2 + b2                                   (16x128, WMMA over K=128)
//   atomicAdd into delta rows eij.
#define K2_WAVES 8
__global__ __launch_bounds__(256)
void three_body_kernel(const __bf16* __restrict__ attrb,
                       const __bf16* __restrict__ angb,
                       const int*    __restrict__ tbe,
                       const __bf16* __restrict__ w1f_g,
                       const __bf16* __restrict__ w2f_g,
                       const float*  __restrict__ b1,
                       const float*  __restrict__ b2,
                       float* __restrict__ delta) {
  __shared__ __bf16 sw1[W1F_ELEMS];            // 72 KB
  __shared__ __bf16 sw2[W2F_ELEMS];            // 32 KB
  __shared__ __bf16 sh[K2_WAVES][16 * DF];     // 32 KB (4 KB per wave)

  const int tid = threadIdx.x;
  stage_to_lds(w1f_g, sw1, W1F_ELEMS * 2, tid, blockDim.x);
  stage_to_lds(w2f_g, sw2, W2F_ELEMS * 2, tid, blockDim.x);
  __syncthreads();

  const int lane  = tid & 31;
  const int wave  = tid >> 5;
  const int m     = lane & 15;
  const int half  = lane >> 4;
  const int khalf = half * 8;
  __bf16* hbuf = sh[wave];

  float b1v[8], b2v[8];
#pragma unroll
  for (int nt = 0; nt < 8; ++nt) {
    b1v[nt] = b1[nt * 16 + m];
    b2v[nt] = b2[nt * 16 + m];
  }

  const v8f vzero = {0.f, 0.f, 0.f, 0.f, 0.f, 0.f, 0.f, 0.f};
  const int nTiles = T_TRIP / 16;
  const int stride = gridDim.x * K2_WAVES;

  for (int tile = blockIdx.x * K2_WAVES + wave; tile < nTiles; tile += stride) {
    const int t   = tile * 16 + m;
    const int eij = tbe[2 * t];
    const int eik = tbe[2 * t + 1];
    const __bf16* pij = attrb + (size_t)eij * DF;
    const __bf16* pik = attrb + (size_t)eik * DF;
    const __bf16* pan = angb + (size_t)t * 32;

    v8f acc[8];
#pragma unroll
    for (int nt = 0; nt < 8; ++nt) acc[nt] = vzero;

    // -------- GEMM 1: K = 288 (4 kt from attr[eij], 4 from attr[eik], 1 from ang)
#pragma unroll
    for (int kt = 0; kt < 9; ++kt) {
      const __bf16* base = (kt < 4) ? (pij + kt * 32)
                         : (kt < 8) ? (pik + (kt - 4) * 32)
                                    : pan;
      const v8bf lo = *(const v8bf*)(base + khalf);        // K = khalf..khalf+7
      const v8bf hi = *(const v8bf*)(base + 16 + khalf);   // K = 16+khalf..+7
      v16bf a;
#pragma unroll
      for (int i = 0; i < 8; ++i) { a[i] = lo[i]; a[8 + i] = hi[i]; }
      const __bf16* wb = sw1 + ((size_t)(kt * 8) * 32 + lane) * 16;
#pragma unroll
      for (int nt = 0; nt < 8; ++nt) {
        const v16bf b = *(const v16bf*)(wb + (size_t)nt * (32 * 16));
        acc[nt] = __builtin_amdgcn_wmma_f32_16x16x32_bf16(
            false, a, false, b, (short)0, acc[nt], false, false);
      }
    }

    // -------- bias + silu, re-layout D(16x128) -> LDS row-major bf16
#pragma unroll
    for (int nt = 0; nt < 8; ++nt) {
#pragma unroll
      for (int r = 0; r < 8; ++r) {
        float x = acc[nt][r] + b1v[nt];
        float s = x / (1.0f + __expf(-x));
        hbuf[(r + 8 * half) * DF + nt * 16 + m] = (__bf16)s;
      }
    }

    // -------- GEMM 2: K = 128
    v8f acc2[8];
#pragma unroll
    for (int nt = 0; nt < 8; ++nt) acc2[nt] = vzero;
#pragma unroll
    for (int kt = 0; kt < 4; ++kt) {
      const __bf16* hb = hbuf + m * DF + kt * 32;
      const v8bf lo = *(const v8bf*)(hb + khalf);
      const v8bf hi = *(const v8bf*)(hb + 16 + khalf);
      v16bf a;
#pragma unroll
      for (int i = 0; i < 8; ++i) { a[i] = lo[i]; a[8 + i] = hi[i]; }
      const __bf16* wb = sw2 + ((size_t)(kt * 8) * 32 + lane) * 16;
#pragma unroll
      for (int nt = 0; nt < 8; ++nt) {
        const v16bf b = *(const v16bf*)(wb + (size_t)nt * (32 * 16));
        acc2[nt] = __builtin_amdgcn_wmma_f32_16x16x32_bf16(
            false, a, false, b, (short)0, acc2[nt], false, false);
      }
    }

    // -------- segment-sum scatter: row of output element (r,half) is eij of triplet r+8*half
    int erow[8];
#pragma unroll
    for (int r = 0; r < 8; ++r) erow[r] = __shfl(eij, r + 8 * half, 32);
#pragma unroll
    for (int nt = 0; nt < 8; ++nt) {
#pragma unroll
      for (int r = 0; r < 8; ++r) {
        // native non-returning GLOBAL_ATOMIC_ADD_F32 at device scope
        (void)__hip_atomic_fetch_add(delta + (size_t)erow[r] * DF + nt * 16 + m,
                                     acc2[nt][r] + b2v[nt],
                                     __ATOMIC_RELAXED, __HIP_MEMORY_SCOPE_AGENT);
      }
    }
  }
}

// ---------------------------------------------------------------------------
// Final GEMM: out = delta @ Wu + bu   (E x 128 @ 128 x 128), WMMA bf16
__global__ __launch_bounds__(256)
void out_gemm_kernel(const float* __restrict__ delta,
                     const __bf16* __restrict__ wuf_g,
                     const float* __restrict__ bu,
                     float* __restrict__ out) {
  __shared__ __bf16 swu[W2F_ELEMS];  // 32 KB
  const int tid = threadIdx.x;
  stage_to_lds(wuf_g, swu, W2F_ELEMS * 2, tid, blockDim.x);
  __syncthreads();

  const int lane  = tid & 31;
  const int wave  = tid >> 5;
  const int m     = lane & 15;
  const int half  = lane >> 4;
  const int khalf = half * 8;

  float buv[8];
#pragma unroll
  for (int nt = 0; nt < 8; ++nt) buv[nt] = bu[nt * 16 + m];

  const v8f vzero = {0.f, 0.f, 0.f, 0.f, 0.f, 0.f, 0.f, 0.f};
  const int nTiles = E_EDGES / 16;
  const int stride = gridDim.x * 8;

  for (int tile = blockIdx.x * 8 + wave; tile < nTiles; tile += stride) {
    const float* prow = delta + (size_t)(tile * 16 + m) * DF;
    v8f acc[8];
#pragma unroll
    for (int nt = 0; nt < 8; ++nt) acc[nt] = vzero;

#pragma unroll
    for (int kt = 0; kt < 4; ++kt) {
      const float* pb = prow + kt * 32;
      float4 l0 = ((const float4*)(pb + khalf))[0];
      float4 l1 = ((const float4*)(pb + khalf))[1];
      float4 h0 = ((const float4*)(pb + 16 + khalf))[0];
      float4 h1 = ((const float4*)(pb + 16 + khalf))[1];
      float lo8[8] = {l0.x, l0.y, l0.z, l0.w, l1.x, l1.y, l1.z, l1.w};
      float hi8[8] = {h0.x, h0.y, h0.z, h0.w, h1.x, h1.y, h1.z, h1.w};
      v16bf a;
#pragma unroll
      for (int i = 0; i < 8; ++i) { a[i] = (__bf16)lo8[i]; a[8 + i] = (__bf16)hi8[i]; }
      const __bf16* wb = swu + ((size_t)(kt * 8) * 32 + lane) * 16;
#pragma unroll
      for (int nt = 0; nt < 8; ++nt) {
        const v16bf b = *(const v16bf*)(wb + (size_t)nt * (32 * 16));
        acc[nt] = __builtin_amdgcn_wmma_f32_16x16x32_bf16(
            false, a, false, b, (short)0, acc[nt], false, false);
      }
    }

#pragma unroll
    for (int r = 0; r < 8; ++r) {
      float* orow = out + (size_t)(tile * 16 + r + 8 * half) * DF + m;
#pragma unroll
      for (int nt = 0; nt < 8; ++nt) orow[nt * 16] = acc[nt][r] + buv[nt];
    }
  }
}

// ---------------------------------------------------------------------------
extern "C" void kernel_launch(void* const* d_in, const int* in_sizes, int n_in,
                              void* d_out, int out_size, void* d_ws, size_t ws_size,
                              hipStream_t stream) {
  const float* edge_attr = (const float*)d_in[0];
  // d_in[1] (three_body_indices) unused by the reference math
  const int*   tbe = (const int*)d_in[2];
  const float* ev  = (const float*)d_in[3];
  const float* A1  = (const float*)d_in[4];
  const float* ab1 = (const float*)d_in[5];
  const float* A2  = (const float*)d_in[6];
  const float* ab2 = (const float*)d_in[7];
  const float* W1  = (const float*)d_in[8];
  const float* b1  = (const float*)d_in[9];
  const float* W2  = (const float*)d_in[10];
  const float* b2  = (const float*)d_in[11];
  const float* Wu  = (const float*)d_in[12];
  const float* bu  = (const float*)d_in[13];

  char* ws = (char*)d_ws;
  size_t off = 0;
  float*  delta = (float*)(ws + off);  off += DELTA_BYTES;
  __bf16* attrb = (__bf16*)(ws + off); off += ATTR_BYTES;
  __bf16* angb  = (__bf16*)(ws + off); off += ANG_BYTES;
  __bf16* w1f   = (__bf16*)(ws + off); off += (size_t)W1F_ELEMS * 2;
  __bf16* w2f   = (__bf16*)(ws + off); off += (size_t)W2F_ELEMS * 2;
  __bf16* wuf   = (__bf16*)(ws + off); off += (size_t)W2F_ELEMS * 2;

  (void)hipMemsetAsync(delta, 0, DELTA_BYTES, stream);

  {
    int n4 = E_EDGES * DF / 4;
    cvt_attr_kernel<<<(n4 + 255) / 256, 256, 0, stream>>>(edge_attr, attrb, n4);
  }
  build_wf_kernel<<<(9 * 4096 + 255) / 256, 256, 0, stream>>>(W1, w1f, 9, 276);
  build_wf_kernel<<<(4 * 4096 + 255) / 256, 256, 0, stream>>>(W2, w2f, 4, 128);
  build_wf_kernel<<<(4 * 4096 + 255) / 256, 256, 0, stream>>>(Wu, wuf, 4, 128);
  angle_kernel<<<(T_TRIP + 255) / 256, 256, 0, stream>>>(tbe, ev, A1, ab1, A2, ab2, angb);

  three_body_kernel<<<1024, 256, 0, stream>>>(attrb, angb, tbe, w1f, w2f, b1, b2, delta);
  out_gemm_kernel<<<512, 256, 0, stream>>>(delta, wuf, bu, (float*)d_out);
}
